// E59cMixedHighwayCell_2147483648042
// MI455X (gfx1250) — matrix operand order repaired and checked by reading, not compiled, not run on hardware
//
#include <hip/hip_runtime.h>
#include <hip/hip_bf16.h>

// ---------------------------------------------------------------- constants
static constexpr int kT  = 512;
static constexpr int kB  = 32;
static constexpr int kD  = 1024;
static constexpr int kTB = kT * kB;           // 16384 GEMM rows
static constexpr int kBD = kB * kD;           // 32768 elems per timestep
static constexpr int kNWG = 32;               // persistent scan workgroups
// LDS row layout for the W_h slice: 1024 bf16 (2048B) + 8B pad inserted after
// each 1024B (TDM pad_interval=7, pad_amount=1) -> 2064B row stride.
// 2064B = 516 dwords == 4 (mod 64 banks): 16 lanes hit 16 disjoint 4-bank groups.
static constexpr int kRowBytes = 2064;
static constexpr unsigned kSmemBytes = kB * kRowBytes + 2 * kB * 4; // 66304

// ---------------------------------------------------------------- types
typedef __bf16          v16bf  __attribute__((ext_vector_type(16)));
typedef float           v8f    __attribute__((ext_vector_type(8)));
typedef unsigned short  v16us  __attribute__((ext_vector_type(16)));
typedef unsigned int    v4u    __attribute__((ext_vector_type(4)));
typedef int             v4i    __attribute__((ext_vector_type(4)));
typedef int             v8i    __attribute__((ext_vector_type(8)));

struct Bits2x128 { uint4 lo, hi; };
struct Bits4x64  { unsigned long long a, b, c, d; };

__device__ __forceinline__ v16bf as_v16bf(uint4 lo, uint4 hi) {
  Bits2x128 t{lo, hi};
  return __builtin_bit_cast(v16bf, t);
}
__device__ __forceinline__ v16bf as_v16bf4(unsigned long long a, unsigned long long b,
                                           unsigned long long c, unsigned long long d) {
  Bits4x64 t{a, b, c, d};
  return __builtin_bit_cast(v16bf, t);
}

// float -> bf16 (round-to-nearest-even)
__device__ __forceinline__ unsigned short f2bf(float f) {
  unsigned u = __builtin_bit_cast(unsigned, f);
  u += 0x7FFFu + ((u >> 16) & 1u);
  return (unsigned short)(u >> 16);
}

__device__ __forceinline__ v16bf pack16(const float* f) {
  v16us t;
#pragma unroll
  for (int i = 0; i < 16; ++i) t[i] = f2bf(f[i]);
  return __builtin_bit_cast(v16bf, t);
}

__device__ __forceinline__ v8f wmma_bf16(v16bf a, v16bf b, v8f c) {
  // D = A(16x32 bf16) * B(32x16 bf16) + C(16x16 f32)
  return __builtin_amdgcn_wmma_f32_16x16x32_bf16(false, a, false, b,
                                                 (short)0, c, false, false);
}

// device-scope sense-reversing barrier across kNWG blocks
__device__ __forceinline__ void gsync(unsigned* cnt, unsigned* gen) {
  __syncthreads();
  if (threadIdx.x == 0) {
    __threadfence();  // release: make this WG's stores device-visible
    unsigned g = __hip_atomic_load(gen, __ATOMIC_RELAXED, __HIP_MEMORY_SCOPE_AGENT);
    unsigned arrived =
        __hip_atomic_fetch_add(cnt, 1u, __ATOMIC_ACQ_REL, __HIP_MEMORY_SCOPE_AGENT);
    if (arrived == (unsigned)kNWG - 1u) {
      __hip_atomic_store(cnt, 0u, __ATOMIC_RELAXED, __HIP_MEMORY_SCOPE_AGENT);
      __hip_atomic_fetch_add(gen, 1u, __ATOMIC_ACQ_REL, __HIP_MEMORY_SCOPE_AGENT);
    } else {
      while (__hip_atomic_load(gen, __ATOMIC_ACQUIRE, __HIP_MEMORY_SCOPE_AGENT) == g)
        __builtin_amdgcn_s_sleep(1);
    }
    __threadfence();  // acquire: invalidate WGP-local cache before re-reading
  }
  __syncthreads();
}

// ---------------------------------------------------------------- kernel 0
__global__ void hw_init_kernel(const float* __restrict__ W,
                               const float* __restrict__ Wh,
                               const float* __restrict__ h0,
                               const float* __restrict__ lb,
                               unsigned short* __restrict__ Wb,
                               unsigned short* __restrict__ Whb,
                               float* __restrict__ hf,
                               unsigned short* __restrict__ hb,
                               float* __restrict__ partial,
                               unsigned* __restrict__ bar,
                               float* __restrict__ hseq0) {
  const int i = blockIdx.x * blockDim.x + threadIdx.x;
  const float beta = 0.1f / (1.0f + __expf(-lb[0]));
  if (i < kD * kD) {
    Wb[i]  = f2bf(W[i]);
    Whb[i] = f2bf(beta * Wh[i]);
  }
  if (i < kBD) {
    float h = h0[i];
    hf[i] = h;
    hb[i] = f2bf(h);
    hseq0[i] = h;             // h output row 0
  }
  if (i < kNWG * kB) partial[i] = 0.0f;
  if (i == 0) { bar[0] = 0u; bar[1] = 0u; }
}

// ---------------------------------------------------------------- kernel 1
// Wxp[m,n] = alpha * (sum_k x[m,k] * W[n,k] + b[n]),  m in [0,16384), n in [0,1024)
__global__ void hw_gemm_wx_kernel(const float* __restrict__ x,
                                  const float* __restrict__ bvec,
                                  const float* __restrict__ la,
                                  const unsigned short* __restrict__ Wb,
                                  float* __restrict__ Wxp) {
  const int tid  = threadIdx.x;
  const int wave = tid >> 5;
  const int lane = tid & 31;
  const int half = (lane < 16) ? 0 : 1;
  const int ln16 = lane & 15;

  const int mtile = blockIdx.x * 8 + wave;     // 16-row tile index
  const int rowA  = mtile * 16 + ln16;         // this lane's A row
  const int nbase = blockIdx.y * 4;            // 4 consecutive 16-col tiles

  v8f acc[4];
#pragma unroll
  for (int j = 0; j < 4; ++j) acc[j] = v8f{0.f, 0.f, 0.f, 0.f, 0.f, 0.f, 0.f, 0.f};

  const float* xrow = x + (long)rowA * kD;

  for (int k = 0; k < kD; k += 32) {
    // A fragment: K = k + half*16 .. +15 (contiguous)
    float fa[16];
    const float4* p0 = (const float4*)(xrow + k + half * 16);
    const float4* p1 = (const float4*)(xrow + k + 8 + half * 16);
    float4 a0 = p0[0], a1 = p0[1], a2 = p1[0], a3 = p1[1];
    fa[0]=a0.x; fa[1]=a0.y; fa[2]=a0.z; fa[3]=a0.w;
    fa[4]=a1.x; fa[5]=a1.y; fa[6]=a1.z; fa[7]=a1.w;
    fa[8]=a2.x; fa[9]=a2.y; fa[10]=a2.z; fa[11]=a2.w;
    fa[12]=a3.x; fa[13]=a3.y; fa[14]=a3.z; fa[15]=a3.w;
    v16bf A = pack16(fa);

#pragma unroll
    for (int j = 0; j < 4; ++j) {
      const int n = (nbase + j) * 16 + ln16;   // B column, K-contiguous in memory
      const uint4* bp = (const uint4*)(Wb + (long)n * kD + k + half * 16);
      v16bf B = as_v16bf(bp[0], bp[1]);
      acc[j] = wmma_bf16(A, B, acc[j]);
    }
  }

  const float alpha = __expf(la[0]);
#pragma unroll
  for (int j = 0; j < 4; ++j) {
    const int c = (nbase + j) * 16 + ln16;
    const float bc = bvec[c];
#pragma unroll
    for (int e = 0; e < 8; ++e) {
      const int r = mtile * 16 + e + half * 8;
      Wxp[(long)r * kD + c] = alpha * (acc[j][e] + bc);
    }
  }
}

// ---------------------------------------------------------------- kernel 2
// Persistent scan: 32 blocks x 128 threads. Block wg owns columns [wg*32, wg*32+32).
__global__ void hw_scan_kernel(const float* __restrict__ Wxp,
                               const unsigned short* __restrict__ Whb,
                               float* __restrict__ hf,
                               unsigned short* __restrict__ hb,
                               float* __restrict__ partial,
                               unsigned* __restrict__ bar,
                               float* __restrict__ outs,
                               float* __restrict__ hseq) {
  extern __shared__ char smem[];                       // dynamic LDS @ offset 0
  unsigned short* Bs    = (unsigned short*)smem;       // 32 rows x 2064B (padded)
  float*          rowsq = (float*)(smem + kB * kRowBytes);   // [32]
  float*          rms   = rowsq + kB;                        // [32]

  const int wg   = blockIdx.x;
  const int tid  = threadIdx.x;
  const int wave = tid >> 5;        // 0..3
  const int lane = tid & 31;
  const int half = (lane < 16) ? 0 : 1;
  const int ln16 = lane & 15;

  const int m      = wave & 1;              // M tile (rows m*16..m*16+15)
  const int ln     = wave >> 1;             // local N tile 0..1
  const int ntile  = wg * 2 + ln;           // global N tile 0..63
  const int coln   = ntile * 16 + ln16;     // this lane's output column
  const int colLds = ln * 16 + ln16;        // column within LDS slice

  // ---- preload this WG's W_h slice (32 cols x 1024 K, bf16) into LDS
#if __has_builtin(__builtin_amdgcn_tensor_load_to_lds)
  if (wave == 0) {
    // Tensor DMA: 2D tensor 1024(K) x 32(cols) of 2-byte elems, whole tile,
    // LDS padding: 2 dwords (8B) after every 256 dwords (1024B) -> 2064B rows.
    unsigned long long ga =
        (unsigned long long)(const void*)(Whb + (unsigned long long)wg * 32u * kD);
    v4u g0;
    g0[0] = 1u;                                       // count=1, user mode
    g0[1] = 0u;                                       // lds_addr = 0 (Bs @ seg start)
    g0[2] = (unsigned)(ga & 0xFFFFFFFFu);             // global_addr[31:0]
    g0[3] = (unsigned)((ga >> 32) & 0x01FFFFFFu) | (2u << 30);  // addr[56:32]|type=2
    v8i g1;
    g1[0] = (int)((1u << 16)      // data_size = 1 -> 2 bytes
                | (1u << 20)      // pad_enable
                | (7u << 22)      // pad_interval = 7 -> every 256 dwords
                | (1u << 25));    // pad_amount = 1 -> 2 dwords (8B)
    g1[1] = (int)(1024u << 16);   // tensor_dim0 = 1024 (lo16 in [31:16])
    g1[2] = (int)(32u << 16);     // tensor_dim0 hi=0 | tensor_dim1 = 32
    g1[3] = (int)(1024u << 16);   // tensor_dim1 hi=0 | tile_dim0 = 1024
    g1[4] = 32;                   // tile_dim1 = 32 | tile_dim2 = 0
    g1[5] = 1024;                 // tensor_dim0_stride[31:0] = 1024 elems
    g1[6] = 0;                    // stride0 hi | tensor_dim1_stride lo
    g1[7] = 0;
    v4i gz = {0, 0, 0, 0};
#if defined(__clang_major__) && (__clang_major__ >= 23)
    v8i gz8 = {0, 0, 0, 0, 0, 0, 0, 0};
    __builtin_amdgcn_tensor_load_to_lds(g0, g1, gz, gz, gz8, 0);
#else
    __builtin_amdgcn_tensor_load_to_lds(g0, g1, gz, gz, 0);
#endif
#if __has_builtin(__builtin_amdgcn_s_wait_tensorcnt)
    __builtin_amdgcn_s_wait_tensorcnt(0);
#else
    asm volatile("s_wait_tensorcnt 0x0" ::: "memory");
#endif
  }
#else
  // Fallback: manual copy into the identical padded layout (32B per chunk).
  for (int c = tid; c < kB * (kD / 16); c += blockDim.x) {
    const int col = c >> 6;                 // 0..31
    const int kc  = (c & 63) * 16;          // element offset 0..1008
    const unsigned long long* src =
        (const unsigned long long*)(Whb + (unsigned long long)(wg * 32 + col) * kD + kc);
    char* dst = (char*)Bs + col * kRowBytes + kc * 2 + ((kc >= 512) ? 8 : 0);
    ((unsigned long long*)dst)[0] = src[0];
    ((unsigned long long*)dst)[1] = src[1];
    ((unsigned long long*)dst)[2] = src[2];
    ((unsigned long long*)dst)[3] = src[3];
  }
#endif
  __syncthreads();

  unsigned* cnt = bar;
  unsigned* gen = bar + 1;

  for (int t = 0; t < kT; ++t) {
    if (tid < kB) rowsq[tid] = 0.0f;

    // ---- GEMM tile: acc = h_prev(bf16) @ (beta*W_h)^T for this 16x16 tile
    v8f acc = v8f{0.f, 0.f, 0.f, 0.f, 0.f, 0.f, 0.f, 0.f};
    const int arow = m * 16 + ln16;
#pragma unroll 4
    for (int k = 0; k < kD; k += 32) {
      const uint4* ap = (const uint4*)(hb + (long)arow * kD + k + half * 16);
      uint4 a0 = ap[0];
      uint4 a1 = *(const uint4*)(hb + (long)arow * kD + k + 8 + half * 16);
      v16bf A = as_v16bf(a0, a1);
      const int kx = k + half * 16;                 // 16-elem chunk start
      const unsigned long long* bq = (const unsigned long long*)
          ((const char*)Bs + colLds * kRowBytes + kx * 2 + ((kx >= 512) ? 8 : 0));
      v16bf B = as_v16bf4(bq[0], bq[1], bq[2], bq[3]);
      acc = wmma_bf16(A, B, acc);
    }

    // ---- pre-activation h + partial sum of squares (16-lane shfl tree)
    float pre[8];
    const float* wx = Wxp + (long)t * kBD;
    __syncthreads();   // rowsq zeroed before atomics
#pragma unroll
    for (int e = 0; e < 8; ++e) {
      const int r = m * 16 + e + half * 8;
      pre[e] = hf[(long)r * kD + coln] + wx[(long)r * kD + coln] + acc[e];
      float v = pre[e] * pre[e];
      v += __shfl_xor(v, 1);
      v += __shfl_xor(v, 2);
      v += __shfl_xor(v, 4);
      v += __shfl_xor(v, 8);
      if (ln16 == 0) atomicAdd(&rowsq[r], v);   // exactly 2 adds/row: deterministic
    }
    __syncthreads();
    if (tid < kB) partial[wg * kB + tid] = rowsq[tid];

    gsync(cnt, gen);   // barrier 1: partial sums visible everywhere

    // ---- fixed-order cross-WG reduction -> rms per row (deterministic)
    if (tid < kB) {
      float s = 0.0f;
      for (int i = 0; i < kNWG; ++i) s += partial[i * kB + tid];
      rms[tid] = rsqrtf(s * (1.0f / kD) + 1e-6f);
    }
    __syncthreads();

    // ---- normalize, activation, write state + outputs
#pragma unroll
    for (int e = 0; e < 8; ++e) {
      const int r = m * 16 + e + half * 8;
      const float h = pre[e] * rms[r];
      const float sg = 1.0f / (1.0f + __expf(-h));
      const long idx = (long)r * kD + coln;
      outs[(long)t * kBD + idx]       = h * h * sg;   // h * silu(h)
      hseq[(long)(t + 1) * kBD + idx] = h;
      hf[idx] = h;
      hb[idx] = f2bf(h);
    }
    if (t + 1 < kT)
      __builtin_prefetch(Wxp + (long)(t + 1) * kBD + (long)(m * 16 + ln16) * kD + ntile * 16, 0, 1);

    gsync(cnt, gen);   // barrier 2: h state visible before next step's GEMM
  }
}

// ---------------------------------------------------------------- launch
extern "C" void kernel_launch(void* const* d_in, const int* in_sizes, int n_in,
                              void* d_out, int out_size, void* d_ws, size_t ws_size,
                              hipStream_t stream) {
  (void)in_sizes; (void)n_in; (void)out_size; (void)ws_size;

  const float* x  = (const float*)d_in[0];
  const float* h0 = (const float*)d_in[1];
  const float* W  = (const float*)d_in[2];
  const float* Wh = (const float*)d_in[3];
  const float* bv = (const float*)d_in[4];
  const float* la = (const float*)d_in[5];
  const float* lb = (const float*)d_in[6];

  float* outs = (float*)d_out;                 // [512, 32, 1024]
  float* hseq = outs + (long)kT * kBD;         // [513, 32, 1024]

  char* ws = (char*)d_ws;
  float*          Wxp     = (float*)ws;                                   // 67108864 B
  unsigned short* Wb      = (unsigned short*)(ws + 67108864);             //  2097152 B
  unsigned short* Whb     = Wb + (long)kD * kD;                           //  2097152 B
  float*          hf      = (float*)(ws + 67108864 + 4194304);            //   131072 B
  unsigned short* hb      = (unsigned short*)((char*)hf + 131072);        //    65536 B
  float*          partial = (float*)((char*)hb + 65536);                  //     4096 B
  unsigned*       bar     = (unsigned*)((char*)partial + 4096);           //        8 B

  hw_init_kernel<<<4096, 256, 0, stream>>>(W, Wh, h0, lb, Wb, Whb, hf, hb,
                                           partial, bar, hseq);

  dim3 g1(kTB / 128, kD / 64);                 // 128 x 16 blocks
  hw_gemm_wx_kernel<<<g1, 256, 0, stream>>>(x, bv, la, Wb, Wxp);

  hw_scan_kernel<<<kNWG, 128, kSmemBytes, stream>>>(Wxp, Whb, hf, hb,
                                                    partial, bar, outs, hseq);
}
